// VectorQuantizer_67808943669361
// MI455X (gfx1250) — matrix operand order repaired and checked by reading, not compiled, not run on hardware
//
#include <hip/hip_runtime.h>

// CDNA5 / gfx1250, wave32. WMMA bf16 + TDM double-buffered B staging.
typedef __attribute__((ext_vector_type(16))) __bf16 v16bf;
typedef __attribute__((ext_vector_type(8)))  float  v8f;
typedef __attribute__((ext_vector_type(4)))  unsigned int u32x4;
typedef __attribute__((ext_vector_type(8)))  int    i32x8;
typedef __attribute__((ext_vector_type(4)))  int    i32x4;

#define B_DIM    64
#define D_DIM    256
#define H_DIM    32
#define W_DIM    32
#define K_CODES  512
#define HW       (H_DIM * W_DIM)            // 1024
#define DHW      (D_DIM * HW)               // 262144
#define N_TOTAL  (B_DIM * HW)               // 65536
#define N_TILES  (N_TOTAL / 16)             // 4096
#define ZQ_ELEMS (B_DIM * D_DIM * HW)       // 16777216
#define IDX_OFF  ZQ_ELEMS
#define LOSS_OFF (ZQ_ELEMS + N_TOTAL)
#define EPSF     1e-12f
#define ZSTRIDE  260                        // 256 + 4 pad: conflict-free LDS rows
#define SSTRIDE  516                        // 512 + 4 pad: conflict-free sim rows
#define CHUNK_BYTES 32768                   // one K-chunk of packed B: 32 nt * 1KB

// ---- TDM: async DMA one contiguous 32KB chunk (8192 dwords) into LDS -------
// D# per CDNA5 ISA ch.8: group0 = {count/flags, lds_addr, global_addr, type},
// group1 = dims/strides. 2D-degenerate tile: tile_dim0=8192 dw, tile_dim1=1.
__device__ __forceinline__ void tdm_load_32k(const void* gsrc, void* lds_dst) {
    unsigned long long ga = (unsigned long long)gsrc;
    unsigned lds_off = (unsigned)(unsigned long long)lds_dst; // addrspace(3) offset
    u32x4 g0;
    g0[0] = 1u;                                            // count=1 (valid user D#)
    g0[1] = lds_off;                                       // lds_addr
    g0[2] = (unsigned)ga;                                  // global_addr[31:0]
    g0[3] = (unsigned)((ga >> 32) & 0x01FFFFFFull)         // global_addr[56:32]
            | 0x80000000u;                                 // type=2 ("image")
    i32x8 g1;
    g1[0] = (2 << 16);          // data_size=4B; wg_mask=0; no pad/iterate/barrier
    g1[1] = (int)(8192u << 16); // tensor_dim0 = 8192 (lo16 at bits 63:48)
    g1[2] = (1 << 16);          // tensor_dim0 hi = 0; tensor_dim1 = 1
    g1[3] = (int)(8192u << 16); // tensor_dim1 hi = 0; tile_dim0 = 8192
    g1[4] = 1;                  // tile_dim1 = 1; tile_dim2 = 0
    g1[5] = 8192;               // tensor_dim0_stride lo
    g1[6] = 0;                  // stride hi / tensor_dim1_stride lo
    g1[7] = 0;
    i32x4 z4 = {0, 0, 0, 0};    // groups 2/3 unused (<=2D)
#if __clang_major__ >= 23
    i32x8 z8 = {0, 0, 0, 0, 0, 0, 0, 0};
    __builtin_amdgcn_tensor_load_to_lds(g0, g1, z4, z4, z8, 0);
#else
    __builtin_amdgcn_tensor_load_to_lds(g0, g1, z4, z4, 0);
#endif
}

// ---------------- Kernel 1: L2-normalize codebook rows (f32) ----------------
__global__ void vq_cb_normalize(const float* __restrict__ cb,
                                float* __restrict__ cbn) {
    __shared__ float red[256];
    const int k = blockIdx.x;
    const int t = threadIdx.x;
    const float v = cb[k * D_DIM + t];
    red[t] = v * v;
    __syncthreads();
    for (int s = 128; s > 0; s >>= 1) {
        if (t < s) red[t] += red[t + s];
        __syncthreads();
    }
    const float scale = 1.0f / fmaxf(sqrtf(red[0]), EPSF);
    cbn[k * D_DIM + t] = v * scale;
}

// -- Kernel 2: pack codebook into K-chunk-major WMMA B-fragment layout ------
// Chunk kk is contiguous 32KB: fragment nt, lane L, elem i holds
//   B[K = kk*32 + (i/8)*16 + (L>=16?8:0) + (i&7)][N = nt*16 + (L&15)]
// flat = ((kk*32 + nt)*32 + L)*16 + i
__global__ void vq_cb_pack(const float* __restrict__ cbn,
                           __bf16* __restrict__ frag) {
    const int f  = blockIdx.x * 256 + threadIdx.x;   // 131072 total
    const int i  = f & 15;
    const int L  = (f >> 4) & 31;
    const int nt = (f >> 9) & 31;
    const int kk = (f >> 14) & 7;
    const int code = nt * 16 + (L & 15);
    const int d    = kk * 32 + (i >> 3) * 16 + ((L >= 16) ? 8 : 0) + (i & 7);
    frag[f] = (__bf16)cbn[code * D_DIM + d];
}

// ---------------- Kernel 3: main VQ (TDM pipeline + WMMA + argmax) ----------
__global__ void __launch_bounds__(256)
vq_main(const float* __restrict__ z_e,
        const float* __restrict__ cbn,
        const unsigned char* __restrict__ bfrag,   // packed bf16 chunks
        float* __restrict__ out,
        float* __restrict__ partials) {
    __shared__ float zraw[16 * ZSTRIDE];                     // raw z tile [row][d]
    __shared__ float sim[16 * SSTRIDE];                      // sims [row][code]
    __shared__ __align__(64) unsigned char bstage[2][CHUNK_BYTES]; // B double buffer
    __shared__ float pss[16][16];
    __shared__ float rowScale[16];
    __shared__ float pmax[16][16];
    __shared__ int   pidx[16][16];
    __shared__ int   bestIdx[16];
    __shared__ float lossRed[256];

    const int T    = blockIdx.x;
    const int t    = threadIdx.x;
    const int lane = t & 31;
    const int wave = t >> 5;
    const int hi   = (lane >= 16) ? 1 : 0;
    const int M    = lane & 15;

    const int w0 = (T & 1) * 16;
    const int h  = (T >> 1) & (H_DIM - 1);
    const int b  = T >> 6;
    const int zbase = b * DHW + h * W_DIM + w0;

    const int r    = t & 15;
    const int dsub = t >> 4;

    // Kick off DMA of B chunk 0 immediately; overlaps the z-tile staging.
    if (wave == 0) tdm_load_32k(bfrag, &bstage[0][0]);

    // Stage strided [B,D,H,W] gather into LDS: 16 rows x 256 d
    #pragma unroll 4
    for (int it = 0; it < 16; ++it) {
        const int d = it * 16 + dsub;
        zraw[r * ZSTRIDE + d] = z_e[zbase + d * HW + r];
    }
    __syncthreads();

    // Row L2 norms -> scale = 1/max(||z||, eps)
    {
        float ss = 0.f;
        #pragma unroll
        for (int dd = 0; dd < 16; ++dd) {
            const float v = zraw[r * ZSTRIDE + dsub * 16 + dd];
            ss += v * v;
        }
        pss[r][dsub] = ss;
    }
    __syncthreads();
    if (t < 16) {
        float ss = 0.f;
        #pragma unroll
        for (int s = 0; s < 16; ++s) ss += pss[t][s];
        rowScale[t] = 1.0f / fmaxf(sqrtf(ss), EPSF);
    }
    if (wave == 0) __builtin_amdgcn_s_wait_tensorcnt(0);  // chunk 0 landed
    __syncthreads();   // rowScale visible + B chunk 0 ready for all waves

    // Build A fragments (ISA 16-bit A 16x32 layout):
    // lane L elem i: M = L&15, K = (i/8)*16 + (L>=16?8:0) + (i&7)
    v16bf afrag[8];
    {
        const float sc = rowScale[M];
        #pragma unroll
        for (int kk = 0; kk < 8; ++kk) {
            v16bf a;
            #pragma unroll
            for (int i = 0; i < 16; ++i) {
                const int d = kk * 32 + (i >> 3) * 16 + hi * 8 + (i & 7);
                a[i] = (__bf16)(zraw[M * ZSTRIDE + d] * sc);
            }
            afrag[kk] = a;
        }
    }

    // Double-buffered K loop: TDM prefetch chunk kk+1 while WMMA on chunk kk.
    v8f c[4] = {v8f{}, v8f{}, v8f{}, v8f{}};
    for (int kk = 0; kk < 8; ++kk) {
        if (wave == 0 && kk < 7)
            tdm_load_32k(bfrag + (size_t)(kk + 1) * CHUNK_BYTES,
                         &bstage[(kk + 1) & 1][0]);
        const v16bf* bp = (const v16bf*)&bstage[kk & 1][0];
        #pragma unroll
        for (int j = 0; j < 4; ++j) {
            const v16bf bv = bp[(wave * 4 + j) * 32 + lane];
            c[j] = __builtin_amdgcn_wmma_f32_16x16x32_bf16(
                       false, afrag[kk], false, bv, (short)0, c[j],
                       false, false);
        }
        if (kk < 7) {
            if (wave == 0) __builtin_amdgcn_s_wait_tensorcnt(0);
            __syncthreads();   // chunk kk reads done; chunk kk+1 visible
        }
    }

    // C layout: elem v -> row = v + 8*hi, col = nt*16 + (lane&15)
    #pragma unroll
    for (int j = 0; j < 4; ++j) {
        const int nt = wave * 4 + j;
        #pragma unroll
        for (int v = 0; v < 8; ++v)
            sim[(v + 8 * hi) * SSTRIDE + nt * 16 + M] = c[j][v];
    }
    __syncthreads();

    // First-match argmax over 512 codes per row (jnp.argmax tie rule)
    {
        const int rr  = t >> 4;
        const int seg = t & 15;
        float mx = -3.402823466e38f;
        int   mi = 0;
        #pragma unroll 8
        for (int ci = seg * 32; ci < seg * 32 + 32; ++ci) {
            const float s = sim[rr * SSTRIDE + ci];
            if (s > mx) { mx = s; mi = ci; }
        }
        pmax[rr][seg] = mx;
        pidx[rr][seg] = mi;
    }
    __syncthreads();
    if (t < 16) {
        float mx = pmax[t][0];
        int   mi = pidx[t][0];
        #pragma unroll
        for (int s = 1; s < 16; ++s)
            if (pmax[t][s] > mx) { mx = pmax[t][s]; mi = pidx[t][s]; }
        bestIdx[t] = mi;
        out[IDX_OFF + T * 16 + t] = (float)mi;   // indices as f32
    }
    __syncthreads();

    // z_q_st == z_q (forward). Write [B,D,H,W]; accumulate commitment partial.
    float acc = 0.f;
    {
        const int ci = bestIdx[r];
        #pragma unroll 4
        for (int dd = 0; dd < 16; ++dd) {
            const int d = dsub * 16 + dd;
            const float q  = cbn[ci * D_DIM + d];
            const float df = zraw[r * ZSTRIDE + d] - q;
            acc += df * df;
            out[zbase + d * HW + r] = q;
        }
    }
    lossRed[t] = acc;
    __syncthreads();
    for (int s = 128; s > 0; s >>= 1) {
        if (t < s) lossRed[t] += lossRed[t + s];
        __syncthreads();
    }
    if (t == 0) partials[T] = lossRed[0];
}

// ---------------- Kernel 4: deterministic loss reduction --------------------
__global__ void vq_loss_reduce(const float* __restrict__ partials,
                               float* __restrict__ out) {
    __shared__ float red[256];
    const int t = threadIdx.x;
    float s = 0.f;
    #pragma unroll
    for (int k = 0; k < N_TILES / 256; ++k) s += partials[k * 256 + t];
    red[t] = s;
    __syncthreads();
    for (int w = 128; w > 0; w >>= 1) {
        if (t < w) red[t] += red[t + w];
        __syncthreads();
    }
    if (t == 0)
        out[LOSS_OFF] = 0.25f * red[0] / (float)ZQ_ELEMS;
}

// ---------------------------------------------------------------------------
extern "C" void kernel_launch(void* const* d_in, const int* in_sizes, int n_in,
                              void* d_out, int out_size, void* d_ws, size_t ws_size,
                              hipStream_t stream) {
    const float* z_e = (const float*)d_in[0];   // [64,256,32,32] f32
    const float* cb  = (const float*)d_in[1];   // [512,256] f32
    float* out = (float*)d_out;                 // z_q_st | indices | loss

    // Workspace layout:
    //   cbn      : 512*256 f32   (512 KB)
    //   bfrag    : 512*256 bf16  (256 KB), K-chunk-major WMMA B-fragment order
    //   partials : 4096 f32      (16 KB)
    char* ws = (char*)d_ws;
    float*         cbn      = (float*)ws;
    __bf16*        frag     = (__bf16*)(ws + (size_t)K_CODES * D_DIM * 4);
    unsigned char* fragb    = (unsigned char*)frag;
    float*         partials = (float*)(ws + (size_t)K_CODES * D_DIM * 4
                                          + (size_t)K_CODES * D_DIM * 2);

    vq_cb_normalize<<<K_CODES, 256, 0, stream>>>(cb, cbn);
    vq_cb_pack<<<K_CODES, 256, 0, stream>>>(cbn, frag);
    vq_main<<<N_TILES, 256, 0, stream>>>(z_e, cbn, fragb, out, partials);
    vq_loss_reduce<<<1, 256, 0, stream>>>(partials, out);
}
